// RPN_70643622084951
// MI455X (gfx1250) — matrix operand order repaired and checked by reading, not compiled, not run on hardware
//
#include <hip/hip_runtime.h>
#include <math.h>

typedef __attribute__((ext_vector_type(16))) __bf16 v16bf;
typedef __attribute__((ext_vector_type(8)))  __bf16 v8bf;
typedef __attribute__((ext_vector_type(8)))  float  v8f;

#define CIN   2048
#define MIDC  512
#define HW    50
#define NPOS  2500
#define KTOT  18432          // 2048 * 9
#define NBATCH 2
#define NANCH 22500          // 2500 * 9
#define NPRE  6000
#define NPOST 300
#define SORTN 32768

// d_out flat offsets
#define LOCS_OFF   0
#define SCORES_OFF 180000
#define ROIS_OFF   270000
#define RIDX_OFF   272400
#define ANCH_OFF   273000

// ---------------------------------------------------------------- packing ---
__global__ void pack_weights(const float* __restrict__ w, __bf16* __restrict__ Wb) {
    int t = blockIdx.x * blockDim.x + threadIdx.x;
    if (t >= MIDC * KTOT) return;
    int co = t / KTOT;
    int q  = t - co * KTOT;
    int khkw = q / CIN;
    int ci   = q - khkw * CIN;
    int kh = khkw / 3, kw = khkw % 3;
    // conv1_w layout (512, 2048, 3, 3)
    float v = w[((co * CIN + ci) * 3 + kh) * 3 + kw];
    Wb[t] = (__bf16)v;
}

__global__ void pack_features(const float* __restrict__ f, __bf16* __restrict__ Fp) {
    int t = blockIdx.x * blockDim.x + threadIdx.x;
    const int TOT = NBATCH * 52 * 52 * CIN;
    if (t >= TOT) return;
    int n  = t / (52 * 52 * CIN);
    int q  = t - n * (52 * 52 * CIN);
    int ph = q / (52 * CIN);
    int q2 = q - ph * (52 * CIN);
    int pw = q2 / CIN;
    int ci = q2 - pw * CIN;
    float v = 0.f;
    if (ph >= 1 && ph <= HW && pw >= 1 && pw <= HW)
        v = f[((n * CIN + ci) * HW + (ph - 1)) * HW + (pw - 1)];
    Fp[t] = (__bf16)v;
}

// ------------------------------------------------------------- WMMA conv ----
__device__ inline v8f wmma_bf16(v16bf a, v16bf b, v8f c) {
    return __builtin_amdgcn_wmma_f32_16x16x32_bf16(false, a, false, b, (short)0, c, false, false);
}

// each wave computes a 32(M) x 32(N) tile: 2x2 WMMA fragments, K = 18432
__global__ __launch_bounds__(256) void conv3_wmma(const __bf16* __restrict__ Wb,
                                                  const __bf16* __restrict__ Fp,
                                                  const float*  __restrict__ bias,
                                                  float* __restrict__ xmid) {
    const int MT = MIDC / 32;            // 16
    const int NT = (NPOS + 31) / 32;     // 79
    int wid  = blockIdx.x * (blockDim.x / 32) + (threadIdx.x >> 5);
    if (wid >= NBATCH * MT * NT) return;
    int lane = threadIdx.x & 31;
    int lrow = lane & 15;
    int lhi  = lane >> 4;                // 0 = lanes 0-15, 1 = lanes 16-31

    int n    = wid / (MT * NT);
    int r    = wid - n * (MT * NT);
    int mblk = r / NT;
    int nblk = r - mblk * NT;
    int mbase = mblk * 32;
    int pbase = nblk * 32;

    int co0 = mbase + lrow;
    int co1 = co0 + 16;
    int p0  = pbase + lrow;
    int p1  = p0 + 16;
    int p0c = p0 < NPOS ? p0 : NPOS - 1;
    int p1c = p1 < NPOS ? p1 : NPOS - 1;
    int hh0 = p0c / HW, ww0 = p0c % HW;
    int hh1 = p1c / HW, ww1 = p1c % HW;

    v8f acc00 = {}, acc01 = {}, acc10 = {}, acc11 = {};

    // A fragment: lane holds row co, elems 0..7 = K kk+lhi*8.., elems 8..15 = K kk+16+lhi*8..
    const __bf16* wa0 = Wb + (size_t)co0 * KTOT + lhi * 8;
    const __bf16* wa1 = Wb + (size_t)co1 * KTOT + lhi * 8;

    for (int khkw = 0; khkw < 9; ++khkw) {
        int kh = khkw / 3, kw = khkw % 3;
        const __bf16* fb0 = Fp + (((size_t)n * 52 + hh0 + kh) * 52 + (ww0 + kw)) * CIN + lhi * 8;
        const __bf16* fb1 = Fp + (((size_t)n * 52 + hh1 + kh) * 52 + (ww1 + kw)) * CIN + lhi * 8;
        const __bf16* wk0 = wa0 + khkw * CIN;
        const __bf16* wk1 = wa1 + khkw * CIN;
        #pragma unroll 4
        for (int c = 0; c < CIN; c += 32) {
            v8bf a0lo = *(const v8bf*)(wk0 + c);
            v8bf a0hi = *(const v8bf*)(wk0 + c + 16);
            v8bf a1lo = *(const v8bf*)(wk1 + c);
            v8bf a1hi = *(const v8bf*)(wk1 + c + 16);
            v16bf A0 = __builtin_shufflevector(a0lo, a0hi, 0,1,2,3,4,5,6,7,8,9,10,11,12,13,14,15);
            v16bf A1 = __builtin_shufflevector(a1lo, a1hi, 0,1,2,3,4,5,6,7,8,9,10,11,12,13,14,15);
            // B fragment: lane holds column p, elems 0..7 = K kk+lhi*8.., elems 8..15 = K kk+16+lhi*8..
            v16bf B0 = __builtin_shufflevector(*(const v8bf*)(fb0 + c), *(const v8bf*)(fb0 + c + 16), 0,1,2,3,4,5,6,7,8,9,10,11,12,13,14,15);
            v16bf B1 = __builtin_shufflevector(*(const v8bf*)(fb1 + c), *(const v8bf*)(fb1 + c + 16), 0,1,2,3,4,5,6,7,8,9,10,11,12,13,14,15);
            acc00 = wmma_bf16(A0, B0, acc00);
            acc01 = wmma_bf16(A0, B1, acc01);
            acc10 = wmma_bf16(A1, B0, acc10);
            acc11 = wmma_bf16(A1, B1, acc11);
        }
    }

    // C/D layout: element v of lane l -> M = v + 8*lhi, N = lrow
    #pragma unroll
    for (int v = 0; v < 8; ++v) {
        int cvoff = v + 8 * lhi;
        int cA = mbase + cvoff;           // mi = 0
        int cB = mbase + 16 + cvoff;      // mi = 1
        float bA = bias[cA], bB = bias[cB];
        if (p0 < NPOS) {
            float r0 = acc00[v] + bA;
            float r1 = acc10[v] + bB;
            xmid[((size_t)n * NPOS + p0) * MIDC + cA] = r0 > 0.f ? r0 : 0.f;
            xmid[((size_t)n * NPOS + p0) * MIDC + cB] = r1 > 0.f ? r1 : 0.f;
        }
        if (p1 < NPOS) {
            float r0 = acc01[v] + bA;
            float r1 = acc11[v] + bB;
            xmid[((size_t)n * NPOS + p1) * MIDC + cA] = r0 > 0.f ? r0 : 0.f;
            xmid[((size_t)n * NPOS + p1) * MIDC + cB] = r1 > 0.f ? r1 : 0.f;
        }
    }
}

// ------------------------------------------------------------- 1x1 convs ----
__global__ void conv1x1_kernel(const float* __restrict__ xmid,
                               const float* __restrict__ locw, const float* __restrict__ locb,
                               const float* __restrict__ scw,  const float* __restrict__ scb,
                               float* __restrict__ out) {
    int t = blockIdx.x * blockDim.x + threadIdx.x;
    if (t >= NBATCH * NPOS * 54) return;
    int n  = t / (NPOS * 54);
    int q  = t - n * (NPOS * 54);
    int p  = q / 54;
    int oc = q - p * 54;
    const float* xp = xmid + ((size_t)n * NPOS + p) * MIDC;
    const float* wp;
    float acc;
    if (oc < 36) { wp = locw + oc * MIDC;        acc = locb[oc]; }
    else         { wp = scw  + (oc - 36) * MIDC; acc = scb[oc - 36]; }
    const float4* x4 = (const float4*)xp;
    const float4* w4 = (const float4*)wp;
    #pragma unroll 4
    for (int c = 0; c < MIDC / 4; ++c) {
        float4 a = x4[c], w = w4[c];
        acc += a.x * w.x + a.y * w.y + a.z * w.z + a.w * w.w;
    }
    if (oc < 36) out[LOCS_OFF   + n * 90000 + p * 36 + oc] = acc;
    else         out[SCORES_OFF + n * 45000 + p * 18 + (oc - 36)] = acc;
}

// --------------------------------------------------------------- anchors ----
__global__ void anchors_kernel(float* __restrict__ out) {
    int i = blockIdx.x * blockDim.x + threadIdx.x;
    if (i >= NANCH) return;
    int a = i % 9, k = i / 9;
    int xg = k / HW, yg = k % HW;     // x-major ravel (matches reference meshgrid)
    const float ratios[3] = {0.5f, 1.0f, 2.0f};
    const float scales[3] = {8.f, 16.f, 32.f};
    float rt = ratios[a / 3], sc = scales[a % 3];
    float hh = 16.f * sc * sqrtf(rt);
    float ww = 16.f * sc * sqrtf(1.f / rt);
    float sy = yg * 16.f, sx = xg * 16.f;
    float* o = out + ANCH_OFF + i * 4;
    o[0] = 8.f - hh * 0.5f + sy;
    o[1] = 8.f - ww * 0.5f + sx;
    o[2] = 8.f + hh * 0.5f + sy;
    o[3] = 8.f + ww * 0.5f + sx;
}

__global__ void roiidx_kernel(float* __restrict__ out) {
    int t = blockIdx.x * blockDim.x + threadIdx.x;
    if (t < NBATCH * NPOST) out[RIDX_OFF + t] = (float)(t / NPOST);
}

// ----------------------------------------------------------------- decode ---
__global__ void decode_kernel(const float* __restrict__ out,
                              const int* __restrict__ imh, const int* __restrict__ imw,
                              float* __restrict__ boxes, float* __restrict__ scor) {
    int t = blockIdx.x * blockDim.x + threadIdx.x;
    if (t >= NBATCH * NANCH) return;
    int n = t / NANCH, i = t - n * NANCH;
    const float* loc = out + LOCS_OFF + n * 90000 + i * 4;
    const float* anc = out + ANCH_OFF + i * 4;
    float s0 = out[SCORES_OFF + n * 45000 + i * 2];
    float s1 = out[SCORES_OFF + n * 45000 + i * 2 + 1];
    float fg = 1.f / (1.f + expf(s0 - s1));
    float H = (float)imh[0], W = (float)imw[0];
    float ah = anc[2] - anc[0], aw = anc[3] - anc[1];
    float acy = anc[0] + 0.5f * ah, acx = anc[1] + 0.5f * aw;
    float cy = loc[0] * ah + acy, cx = loc[1] * aw + acx;
    float hh = expf(loc[2]) * ah, ww = expf(loc[3]) * aw;
    float y1 = fminf(fmaxf(cy - 0.5f * hh, 0.f), H);
    float x1 = fminf(fmaxf(cx - 0.5f * ww, 0.f), W);
    float y2 = fminf(fmaxf(cy + 0.5f * hh, 0.f), H);
    float x2 = fminf(fmaxf(cx + 0.5f * ww, 0.f), W);
    bool valid = ((y2 - y1) >= 16.f) && ((x2 - x1) >= 16.f);
    float* b = boxes + (size_t)t * 4;
    b[0] = y1; b[1] = x1; b[2] = y2; b[3] = x2;
    scor[t] = valid ? fg : -__builtin_inff();
}

// ------------------------------------------------------------ sort + NMS ----
__global__ void sortinit_kernel(const float* __restrict__ scor,
                                float* __restrict__ skey, int* __restrict__ sval) {
    int t = blockIdx.x * blockDim.x + threadIdx.x;
    if (t >= NBATCH * SORTN) return;
    int n = t >> 15, r = t & (SORTN - 1);
    if (r < NANCH) { skey[t] = scor[n * NANCH + r]; sval[t] = r; }
    else           { skey[t] = -__builtin_inff();   sval[t] = r; }
}

__global__ void bitonic_pass(float* __restrict__ skey, int* __restrict__ sval, int j, int kk) {
    int i = blockIdx.x * blockDim.x + threadIdx.x;
    if (i >= NBATCH * SORTN) return;
    int t = i & (SORTN - 1);
    int ixj = i ^ j;                       // j < SORTN -> stays in segment
    if (ixj <= i) return;
    bool up = ((t & kk) == 0);
    float s1 = skey[i], s2 = skey[ixj];
    int v1 = sval[i], v2 = sval[ixj];
    bool before_ji = (s2 > s1) || (s2 == s1 && v2 < v1);  // ixj ranks earlier
    bool before_ij = (s1 > s2) || (s1 == s2 && v1 < v2);
    bool sw = up ? before_ji : before_ij;
    if (sw) { skey[i] = s2; skey[ixj] = s1; sval[i] = v2; sval[ixj] = v1; }
}

__global__ void gather_kernel(const float* __restrict__ boxes, const int* __restrict__ sval,
                              float* __restrict__ topb) {
    int t = blockIdx.x * blockDim.x + threadIdx.x;
    if (t >= NBATCH * NPRE) return;
    int n = t / NPRE, r = t - n * NPRE;
    int idx = sval[n * SORTN + r];
    const float* s = boxes + ((size_t)n * NANCH + idx) * 4;
    float* d = topb + (size_t)t * 4;
    d[0] = s[0]; d[1] = s[1]; d[2] = s[2]; d[3] = s[3];
}

__global__ __launch_bounds__(1024) void nms_kernel(const float* __restrict__ skey,
                                                   const float* __restrict__ topb,
                                                   float* __restrict__ out) {
    __shared__ unsigned keepm[(NPRE + 31) / 32];   // 188 words
    int n = blockIdx.x;
    int tid = threadIdx.x, bs = blockDim.x;
    const float* keys = skey + n * SORTN;
    const float* B = topb + (size_t)n * NPRE * 4;
    for (int w = tid; w < (NPRE + 31) / 32; w += bs) {
        unsigned bits = 0;
        for (int b2 = 0; b2 < 32; ++b2) {
            int r = w * 32 + b2;
            if (r < NPRE && keys[r] > -3.0e38f) bits |= (1u << b2);
        }
        keepm[w] = bits;
    }
    for (int t = tid; t < NPOST * 4; t += bs) out[ROIS_OFF + n * NPOST * 4 + t] = 0.f;
    __syncthreads();
    for (int i = 0; i < NPRE; ++i) {
        bool ki = (keepm[i >> 5] >> (i & 31)) & 1;
        if (ki) {
            float iy1 = B[i * 4], ix1 = B[i * 4 + 1], iy2 = B[i * 4 + 2], ix2 = B[i * 4 + 3];
            float ia = (iy2 - iy1) * (ix2 - ix1);
            for (int j = i + 1 + tid; j < NPRE; j += bs) {
                float jy1 = B[j * 4], jx1 = B[j * 4 + 1], jy2 = B[j * 4 + 2], jx2 = B[j * 4 + 3];
                float ty = fmaxf(iy1, jy1), tx = fmaxf(ix1, jx1);
                float by = fminf(iy2, jy2), bx = fminf(ix2, jx2);
                float ih = by - ty, iw = bx - tx;
                float inter = (ih > 0.f ? ih : 0.f) * (iw > 0.f ? iw : 0.f);
                float ja = (jy2 - jy1) * (jx2 - jx1);
                float iou = inter / (ia + ja - inter);
                if (iou > 0.7f) atomicAnd(&keepm[j >> 5], ~(1u << (j & 31)));
            }
        }
        __syncthreads();
    }
    if (tid == 0) {
        int cnt = 0;
        for (int r = 0; r < NPRE && cnt < NPOST; ++r) {
            if ((keepm[r >> 5] >> (r & 31)) & 1) {
                float* o = out + ROIS_OFF + n * NPOST * 4 + cnt * 4;
                o[0] = B[r * 4]; o[1] = B[r * 4 + 1]; o[2] = B[r * 4 + 2]; o[3] = B[r * 4 + 3];
                ++cnt;
            }
        }
    }
}

// ----------------------------------------------------------------- launch ---
extern "C" void kernel_launch(void* const* d_in, const int* in_sizes, int n_in,
                              void* d_out, int out_size, void* d_ws, size_t ws_size,
                              hipStream_t stream) {
    const float* features = (const float*)d_in[0];
    const float* conv1_w  = (const float*)d_in[1];
    const float* conv1_b  = (const float*)d_in[2];
    const float* loc_w    = (const float*)d_in[3];
    const float* loc_b    = (const float*)d_in[4];
    const float* score_w  = (const float*)d_in[5];
    const float* score_b  = (const float*)d_in[6];
    const int*   img_h    = (const int*)d_in[7];
    const int*   img_w    = (const int*)d_in[8];
    float* out = (float*)d_out;

    char* ws = (char*)d_ws;
    __bf16* Wb   = (__bf16*)(ws);                        // 512*18432*2      = 18,874,368
    __bf16* Fp   = (__bf16*)(ws + 18874368);             // 2*52*52*2048*2   = 22,151,168
    float*  xmid = (float*) (ws + 41025536);             // 2*2500*512*4     = 10,240,000
    float*  boxes= (float*) (ws + 51265536);             // 2*22500*4*4      =    720,000
    float*  scor = (float*) (ws + 51985536);             // 2*22500*4        =    180,000
    float*  skey = (float*) (ws + 52165536);             // 2*32768*4        =    262,144
    int*    sval = (int*)   (ws + 52427680);             // 2*32768*4        =    262,144
    float*  topb = (float*) (ws + 52689824);             // 2*6000*4*4       =    192,000

    const int T = 256;
    pack_weights<<<(MIDC * KTOT + T - 1) / T, T, 0, stream>>>(conv1_w, Wb);
    pack_features<<<(NBATCH * 52 * 52 * CIN + T - 1) / T, T, 0, stream>>>(features, Fp);
    anchors_kernel<<<(NANCH + T - 1) / T, T, 0, stream>>>(out);
    roiidx_kernel<<<(NBATCH * NPOST + T - 1) / T, T, 0, stream>>>(out);

    // 2528 waves total, 8 waves / 256-thread block -> 316 blocks
    conv3_wmma<<<316, 256, 0, stream>>>(Wb, Fp, conv1_b, xmid);

    conv1x1_kernel<<<(NBATCH * NPOS * 54 + T - 1) / T, T, 0, stream>>>(
        xmid, loc_w, loc_b, score_w, score_b, out);

    decode_kernel<<<(NBATCH * NANCH + T - 1) / T, T, 0, stream>>>(out, img_h, img_w, boxes, scor);

    sortinit_kernel<<<(NBATCH * SORTN + T - 1) / T, T, 0, stream>>>(scor, skey, sval);
    for (int kk = 2; kk <= SORTN; kk <<= 1)
        for (int j = kk >> 1; j > 0; j >>= 1)
            bitonic_pass<<<(NBATCH * SORTN + T - 1) / T, T, 0, stream>>>(skey, sval, j, kk);

    gather_kernel<<<(NBATCH * NPRE + T - 1) / T, T, 0, stream>>>(boxes, sval, topb);
    nms_kernel<<<NBATCH, 1024, 0, stream>>>(skey, topb, out);
}
